// Topm_CrossAttention_Restormer_Privileged_3882650436784
// MI455X (gfx1250) — matrix operand order repaired and checked by reading, not compiled, not run on hardware
//
#include <hip/hip_runtime.h>
#include <hip/hip_bf16.h>

typedef _Float16 f16;
typedef __attribute__((ext_vector_type(16))) _Float16 v16h;
typedef __attribute__((ext_vector_type(8)))  float    v8f;

#define TOPK 28
#define LDSTRIDE 132   // 128 + 4 dword pad: LDS bank = (4k + n) % 64 -> conflict-lean

// ---------------------------------------------------------------------------
// WMMA fragment loaders (contiguous k = 16*half + s order in both A and B, wave32)
// ---------------------------------------------------------------------------

// A fragment: 16x32 (MxK) f16. A row-major, lda (floats, even). lane%16 = M,
// half = lane/16: V0..7 hold K = half*16+{0..15}.
// K-pairs are contiguous -> float2 loads.
__device__ inline v16h load_a_frag(const float* __restrict__ A, int lda, int lane) {
    int m = lane & 15;
    int half = lane >> 4;
    const float2* row = (const float2*)(A + (size_t)m * lda);
    v16h a;
#pragma unroll
    for (int i = 0; i < 8; ++i) {
        int kk = half * 16 + 2 * i;
        float2 p = row[kk >> 1];
        a[2 * i]     = (f16)p.x;
        a[2 * i + 1] = (f16)p.y;
    }
    return a;
}

// B fragment: 32x16 (KxN) f16 where B[k][n] = M[k*ldm + n] (global, k-strided).
__device__ inline v16h load_b_frag(const float* __restrict__ M, int ldm, int lane) {
    int n = lane & 15;
    int half = lane >> 4;
    v16h b;
#pragma unroll
    for (int i = 0; i < 8; ++i) {
        int k0 = half * 16 + 2 * i;
        b[2 * i]     = (f16)M[(size_t)k0 * ldm + n];
        b[2 * i + 1] = (f16)M[(size_t)(k0 + 1) * ldm + n];
    }
    return b;
}

// B fragment from LDS-staged tile xs[32][LDSTRIDE] (f32), columns ncol..ncol+15.
__device__ inline v16h load_b_frag_lds(const float* xs, int ncol, int lane) {
    int n = ncol + (lane & 15);
    int half = lane >> 4;
    v16h b;
#pragma unroll
    for (int i = 0; i < 8; ++i) {
        int k0 = half * 16 + 2 * i;
        b[2 * i]     = (f16)xs[k0 * LDSTRIDE + n];
        b[2 * i + 1] = (f16)xs[(k0 + 1) * LDSTRIDE + n];
    }
    return b;
}

// B fragment where logical B[k][n] = M[n*ldm + k]: 16 contiguous floats/lane
// -> 4x float4 (global_load_b128).
__device__ inline v16h load_bT_frag(const float* __restrict__ M, int ldm, int lane) {
    int n = lane & 15;
    int half = lane >> 4;
    const float4* row = (const float4*)(M + (size_t)n * ldm + half * 16);
    v16h b;
#pragma unroll
    for (int i = 0; i < 4; ++i) {
        float4 p = row[i];
        b[4 * i]     = (f16)p.x;
        b[4 * i + 1] = (f16)p.y;
        b[4 * i + 2] = (f16)p.z;
        b[4 * i + 3] = (f16)p.w;
    }
    return b;
}

// C/D 16x16 f32 store: VGPR r -> M = r + (lane/16)*8, N = lane%16.
__device__ inline void store_c(float* __restrict__ C, int ldc, int lane, v8f acc, float scale) {
    int n = lane & 15;
    int half = lane >> 4;
#pragma unroll
    for (int r = 0; r < 8; ++r)
        C[(size_t)(r + half * 8) * ldc + n] = acc[r] * scale;
}

// ---------------------------------------------------------------------------
// 1) conv1x1 as GEMM:  Y[b,o,n] = sum_c Wt[o,c] * X[b,c,n]
//    grid = (N/128, O/16, B), block = 256 (8 waves, each a 16x16 N-subtile).
//    X tile (32K x 128N f32) staged to LDS with global_load_async_to_lds_b128
//    (ASYNCcnt-tracked), then B-fragments built from LDS.
// ---------------------------------------------------------------------------
__global__ __launch_bounds__(256) void conv1x1_wmma_kernel(
    const float* __restrict__ X, const float* __restrict__ Wt,
    float* __restrict__ Y, int Cin, int Cout, int N)
{
    __shared__ float xs[32 * LDSTRIDE];
    int t = threadIdx.x;
    int lane = t & 31;
    int wave = t >> 5;
    int b    = blockIdx.z;
    int m0   = blockIdx.y * 16;
    int nblk = blockIdx.x * 128;
    const float* Xb = X + (size_t)b * Cin * N + nblk;

    int krow = t >> 3;        // 0..31  (tile row)
    int ncol = (t & 7) * 16;  // 0..112 (16 dwords per thread, 4x b128)

    v8f acc = {};
    for (int k0 = 0; k0 < Cin; k0 += 32) {
        __syncthreads();   // previous iteration's readers done before overwrite
        const float* gsrc = Xb + (size_t)(k0 + krow) * N + ncol;
        float* ldst = &xs[krow * LDSTRIDE + ncol];
#if defined(__AMDGCN__) && defined(__gfx1250__)
#pragma unroll
        for (int j = 0; j < 4; ++j) {
            unsigned la = (unsigned)(uintptr_t)(ldst + j * 4);   // LDS byte offset
            asm volatile("global_load_async_to_lds_b128 %0, %1, off"
                         :: "v"(la), "v"(gsrc + j * 4) : "memory");
        }
        asm volatile("s_wait_asynccnt 0x0" ::: "memory");
#else
#pragma unroll
        for (int j = 0; j < 16; ++j) ldst[j] = gsrc[j];
#endif
        __syncthreads();
        v16h a  = load_a_frag(Wt + (size_t)m0 * Cin + k0, Cin, lane);
        v16h bf = load_b_frag_lds(xs, wave * 16, lane);
        acc = __builtin_amdgcn_wmma_f32_16x16x32_f16(false, a, false, bf,
                                                     (short)0, acc, false, false);
    }
    store_c(Y + (size_t)b * Cout * N + (size_t)m0 * N + nblk + wave * 16,
            N, lane, acc, 1.0f);
}

// ---------------------------------------------------------------------------
// 2) depthwise 3x3, SAME zero padding (cross-correlation, matches lax.conv)
// ---------------------------------------------------------------------------
__global__ __launch_bounds__(256) void dwconv3_kernel(
    const float* __restrict__ X, const float* __restrict__ Wdw,
    float* __restrict__ Y, int C, int HH, int WW, long total)
{
    long i = (long)blockIdx.x * blockDim.x + threadIdx.x;
    if (i >= total) return;
    int x = (int)(i % WW);
    long t = i / WW;
    int y = (int)(t % HH); t /= HH;
    int c = (int)(t % C);
    const float* base = X + (i - x - (long)y * WW);   // start of (b,c) plane
    const float* wk = Wdw + c * 9;
    float s = 0.f;
#pragma unroll
    for (int ky = 0; ky < 3; ++ky) {
        int yy = y + ky - 1;
        if ((unsigned)yy >= (unsigned)HH) continue;
#pragma unroll
        for (int kx = 0; kx < 3; ++kx) {
            int xx = x + kx - 1;
            if ((unsigned)xx >= (unsigned)WW) continue;
            s += base[(long)yy * WW + xx] * wk[ky * 3 + kx];
        }
    }
    Y[i] = s;
}

// ---------------------------------------------------------------------------
// 3) per-row L2 norm: nrm[b*rows+c] = max(||X[b,c,:]||, 1e-12). One wave/row.
// ---------------------------------------------------------------------------
__global__ __launch_bounds__(256) void rownorm_kernel(
    const float* __restrict__ X, float* __restrict__ nrm,
    int rowsPerBatch, int batchStride, int N)
{
    int wid  = blockIdx.x * (blockDim.x >> 5) + (threadIdx.x >> 5);
    int lane = threadIdx.x & 31;
    int b = wid / rowsPerBatch;
    int c = wid % rowsPerBatch;
    const float* row = X + (size_t)b * batchStride + (size_t)c * N;
    float s = 0.f;
    for (int i = lane * 4; i < N; i += 32 * 4) {
        float4 v = *(const float4*)(row + i);
        s += v.x * v.x + v.y * v.y + v.z * v.z + v.w * v.w;
    }
#pragma unroll
    for (int off = 16; off; off >>= 1) s += __shfl_xor(s, off, 32);
    if (lane == 0) nrm[wid] = fmaxf(sqrtf(s), 1e-12f);
}

// ---------------------------------------------------------------------------
// 4) Gram matrix G[b,h,c,d] = sum_n q[..c,n]*k[..d,n], 16 n-chunks, f32 atomics.
//    grid=(16, HEADS, B), block=128 (4 waves = 2x2 quadrants of the 32x32 tile)
// ---------------------------------------------------------------------------
__global__ __launch_bounds__(128) void attn_gemm_kernel(
    const float* __restrict__ Q,   // [B,192,N]
    const float* __restrict__ K,   // [B,384,N] (k = first 192 channels)
    float* __restrict__ G,         // [B,6,32,32] pre-zeroed
    int N, int nChunks)
{
    int lane = threadIdx.x & 31;
    int wave = threadIdx.x >> 5;
    int mi = (wave >> 1) * 16;
    int di = (wave & 1) * 16;
    int h = blockIdx.y, b = blockIdx.z;
    const float* Qh = Q + ((size_t)b * 192 + h * 32 + mi) * N;
    const float* Kh = K + ((size_t)b * 384 + h * 32 + di) * N;
    int nPer = N / nChunks;
    int nbeg = blockIdx.x * nPer;
    v8f acc = {};
    for (int n0 = nbeg; n0 < nbeg + nPer; n0 += 32) {
        v16h a  = load_a_frag(Qh + n0, N, lane);   // 16 rows c, K = 32 pixels
        v16h bf = load_bT_frag(Kh + n0, N, lane);  // B[k][d] = Kh[d*N + n0 + k]
        acc = __builtin_amdgcn_wmma_f32_16x16x32_f16(false, a, false, bf,
                                                     (short)0, acc, false, false);
    }
    float* Gt = G + (((size_t)b * 6 + h) * 32 + mi) * 32 + di;
    int n = lane & 15, half = lane >> 4;
#pragma unroll
    for (int r = 0; r < 8; ++r)
        atomicAdd(&Gt[(size_t)(r + half * 8) * 32 + n], acc[r]);
}

// ---------------------------------------------------------------------------
// 5) scale by temperature/(|q||k|), keep top-28 of 32, softmax. One wave/row.
// ---------------------------------------------------------------------------
__global__ __launch_bounds__(256) void topk_softmax_kernel(
    const float* __restrict__ G, const float* __restrict__ nq,
    const float* __restrict__ nk, const float* __restrict__ temp,
    float* __restrict__ S)
{
    int lane = threadIdx.x & 31;
    int row = blockIdx.x * (blockDim.x >> 5) + (threadIdx.x >> 5); // [0, B*6*32)
    int c  = row & 31;
    int bh = row >> 5;
    int h = bh % 6, b = bh / 6;
    int d = lane;
    float g  = G[(size_t)row * 32 + d];
    float qn = nq[b * 192 + h * 32 + c];
    float kn = nk[b * 192 + h * 32 + d];
    float v = g / (qn * kn) * temp[h];
    // rank with (value desc, index asc) tie-break -> matches lax.top_k selection
    int cnt = 0;
#pragma unroll
    for (int j = 0; j < 32; ++j) {
        float vj = __shfl(v, j, 32);
        cnt += (vj > v) || (vj == v && j < d);
    }
    bool keep = cnt < TOPK;
    float mv = keep ? v : -__builtin_inff();
    float mx = mv;
#pragma unroll
    for (int off = 16; off; off >>= 1) mx = fmaxf(mx, __shfl_xor(mx, off, 32));
    float e = keep ? __expf(v - mx) : 0.f;
    float sum = e;
#pragma unroll
    for (int off = 16; off; off >>= 1) sum += __shfl_xor(sum, off, 32);
    S[(size_t)row * 32 + d] = e / sum;
}

// ---------------------------------------------------------------------------
// 6) out[b,h,c,n] = attn4 * sum_d S[c,d] * V[b,h,d,n]   (K=32, single WMMA)
//    grid = (N/128, 2, B*HEADS), block = 256
// ---------------------------------------------------------------------------
__global__ __launch_bounds__(256) void attn_out_kernel(
    const float* __restrict__ S,   // [B*6,32,32]
    const float* __restrict__ V,   // [B,384,N], v = channels 192..383
    const float* __restrict__ a4,
    float* __restrict__ O_,        // [B,192,N]
    int N)
{
    int lane = threadIdx.x & 31, wave = threadIdx.x >> 5;
    int h = blockIdx.z % 6, b = blockIdx.z / 6;
    int m0 = blockIdx.y * 16;
    int n0 = blockIdx.x * 128 + wave * 16;
    const float* St = S + ((size_t)(b * 6 + h) * 32 + m0) * 32;
    const float* Vh = V + ((size_t)b * 384 + 192 + h * 32) * N;
    v16h a  = load_a_frag(St, 32, lane);
    v16h bf = load_b_frag(Vh + n0, N, lane);
    v8f acc = {};
    acc = __builtin_amdgcn_wmma_f32_16x16x32_f16(false, a, false, bf,
                                                 (short)0, acc, false, false);
    store_c(O_ + ((size_t)b * 192 + h * 32 + m0) * N + n0, N, lane, acc, a4[0]);
}

// ---------------------------------------------------------------------------
extern "C" void kernel_launch(void* const* d_in, const int* in_sizes, int n_in,
                              void* d_out, int out_size, void* d_ws, size_t ws_size,
                              hipStream_t stream)
{
    (void)in_sizes; (void)n_in; (void)out_size; (void)ws_size;
    const float* x_q     = (const float*)d_in[0];
    const float* x_kv    = (const float*)d_in[1];
    const float* w_q     = (const float*)d_in[2];
    const float* w_q_dw  = (const float*)d_in[3];
    const float* w_kv    = (const float*)d_in[4];
    const float* w_kv_dw = (const float*)d_in[5];
    const float* w_proj  = (const float*)d_in[6];
    const float* temp    = (const float*)d_in[7];
    const float* a4      = (const float*)d_in[8];

    constexpr int B = 8, C = 192, H = 128, W = 128, N = H * W, HEADS = 6;
    float* ws = (float*)d_ws;
    const size_t SZQ = (size_t)B * C * N;
    const size_t SZKV = 2 * SZQ;

    float* qlin  = ws;                 // conv1x1(q); reused later for attn output
    float* kvlin = qlin + SZQ;
    float* qb    = kvlin + SZKV;       // dwconv(q)
    float* kvb   = qb + SZQ;           // dwconv(kv): k = [0..192), v = [192..384)
    float* nq    = kvb + SZKV;
    float* nk    = nq + (size_t)B * C;
    float* G     = nk + (size_t)B * C;                 // [B,6,32,32]
    float* Ssc   = G + (size_t)B * HEADS * 32 * 32;    // softmax scores
    float* ob    = qlin;               // reuse (qlin dead after dwconv)

    dim3 blk(256);

    conv1x1_wmma_kernel<<<dim3(N / 128, C / 16, B), blk, 0, stream>>>(
        x_q, w_q, qlin, C, C, N);
    conv1x1_wmma_kernel<<<dim3(N / 128, 2 * C / 16, B), blk, 0, stream>>>(
        x_kv, w_kv, kvlin, C, 2 * C, N);

    long tq = (long)B * C * N;
    dwconv3_kernel<<<dim3((unsigned)(tq / 256)), blk, 0, stream>>>(
        qlin, w_q_dw, qb, C, H, W, tq);
    long tkv = 2 * tq;
    dwconv3_kernel<<<dim3((unsigned)(tkv / 256)), blk, 0, stream>>>(
        kvlin, w_kv_dw, kvb, 2 * C, H, W, tkv);

    rownorm_kernel<<<dim3(B * C / 8), blk, 0, stream>>>(qb, nq, C, C * N, N);
    rownorm_kernel<<<dim3(B * C / 8), blk, 0, stream>>>(kvb, nk, C, 2 * C * N, N);

    hipMemsetAsync(G, 0, (size_t)B * HEADS * 32 * 32 * sizeof(float), stream);
    attn_gemm_kernel<<<dim3(16, HEADS, B), dim3(128), 0, stream>>>(qb, kvb, G, N, 16);

    topk_softmax_kernel<<<dim3(B * HEADS * 32 / 8), blk, 0, stream>>>(
        G, nq, nk, temp, Ssc);

    attn_out_kernel<<<dim3(N / 128, 2, B * HEADS), blk, 0, stream>>>(
        Ssc, kvb, a4, ob, N);

    conv1x1_wmma_kernel<<<dim3(N / 128, C / 16, B), blk, 0, stream>>>(
        ob, w_proj, (float*)d_out, C, C, N);
}
